// ComplexMixture_56289841382041
// MI455X (gfx1250) — matrix operand order repaired and checked by reading, not compiled, not run on hardware
//
#include <hip/hip_runtime.h>

typedef __attribute__((ext_vector_type(16))) _Float16 v16h;
typedef __attribute__((ext_vector_type(8)))  _Float16 v8h;
typedef __attribute__((ext_vector_type(8)))  float    v8f;
typedef __attribute__((ext_vector_type(4)))  unsigned v4u;
typedef __attribute__((ext_vector_type(8)))  int      v8i;
typedef __attribute__((ext_vector_type(4)))  int      v4i;

#define BATCH 8
#define SEQ   512
#define DIM   768
#define NELEM (BATCH*SEQ*DIM)      // halves per workspace array (3,145,728)

// LDS tile: 64 rows x 32 K-halves + 8 halves pad -> pitch 40 halves (80B,
// 20 dwords, coprime with 64 banks). TDM reproduces the pad natively:
// pad_interval=3 (after 16 dwords), pad_amount=3 (4 dwords).
#define PITCH 40
#define TILEH (64*PITCH)           // 2560 halves = 5120 B per tile

union F16Frag { v16h v; v8h h[2]; };

// Split f32 into f16 hi + f16 lo, packed into one u32 (hi in [15:0]).
__device__ __forceinline__ unsigned pack_split(float x) {
  _Float16 hi = (_Float16)x;
  _Float16 lo = (_Float16)(x - (float)hi);
  unsigned short uh = __builtin_bit_cast(unsigned short, hi);
  unsigned short ul = __builtin_bit_cast(unsigned short, lo);
  return (unsigned)uh | ((unsigned)ul << 16);
}

__device__ __forceinline__ v8f wmma_f16(const F16Frag& a, const F16Frag& b, v8f c) {
  return __builtin_amdgcn_wmma_f32_16x16x32_f16(false, a.v, false, b.v,
                                                (short)0, c, false, false);
}

// ---------------------------------------------------------------------------
// Precompute: build 8 TRANSPOSED half arrays [b][d][s] in d_ws:
//   0:wRhi 1:wRlo 2:wIhi 3:wIlo 4:Rhi 5:Rlo 6:Ihi 7:Ilo
// LDS-tiled transpose: coalesced reads ([s][d]) and coalesced writes ([d][s]).
// ---------------------------------------------------------------------------
__global__ __launch_bounds__(256) void split_transpose_kernel(
    const float* __restrict__ real, const float* __restrict__ imag,
    const float* __restrict__ weight, unsigned* __restrict__ ws)
{
  __shared__ unsigned tp[4*64*35];   // 4 tiles, pitch 35 dwords (odd -> no conflicts)
  const int t  = threadIdx.x;
  const int db = blockIdx.x * 64;
  const int sb = blockIdx.y * 64;
  const int b  = blockIdx.z;
  const int WPA = NELEM/2;           // u32 words per array

  for (int phase = 0; phase < 2; ++phase) {
    const float* src = (phase ? imag : real) + (size_t)b*SEQ*DIM;
    if (phase) __syncthreads();      // drain of phase 0 done before overwrite
#pragma unroll
    for (int p = 0; p < 8; ++p) {
      int idx = t + p*256;           // 0..2047 -> (d, s-pair)
      int d  = idx & 63;
      int sp = idx >> 6;             // 0..31
      int s  = sp*2;
      float f0 = src[(size_t)(sb+s  )*DIM + db + d];
      float f1 = src[(size_t)(sb+s+1)*DIM + db + d];
      float w0 = weight[b*SEQ + sb + s];
      float w1 = weight[b*SEQ + sb + s + 1];
      unsigned u0 = pack_split(f0),      u1 = pack_split(f1);
      unsigned q0 = pack_split(f0 * w0), q1 = pack_split(f1 * w1);
      int o = d*35 + sp;
      tp[0*2240 + o] = (q0 & 0xFFFFu) | (q1 << 16);          // scaled hi (s,s+1)
      tp[1*2240 + o] = (q0 >> 16)     | (q1 & 0xFFFF0000u);  // scaled lo
      tp[2*2240 + o] = (u0 & 0xFFFFu) | (u1 << 16);          // hi
      tp[3*2240 + o] = (u0 >> 16)     | (u1 & 0xFFFF0000u);  // lo
    }
    __syncthreads();
    // drain: scaled pair -> arrays {0,1} or {2,3}; unscaled -> {4,5} or {6,7}
#pragma unroll
    for (int q = 0; q < 4; ++q) {
      const int a = (q < 2) ? (phase*2 + q) : (4 + phase*2 + (q - 2));
      unsigned* dst = ws + (size_t)a * WPA;
#pragma unroll
      for (int p = 0; p < 8; ++p) {
        int idx = t + p*256;         // 0..2047 -> (d, s-word)
        int d  = idx >> 5;
        int sw = idx & 31;
        dst[(size_t)(b*DIM + db + d)*(SEQ/2) + (sb/2) + sw] = tp[q*2240 + d*35 + sw];
      }
    }
  }
}

// ---------------------------------------------------------------------------
// Shared compute step: fragment loads + 24 WMMAs on one staged buffer.
// ---------------------------------------------------------------------------
__device__ __forceinline__ void compute_step(const _Float16* base,
    int wm, int wn, int hs, int lr, v8f accR[2], v8f accIp[2], v8f accIm[2])
{
  const _Float16* aRh = base + 0*TILEH;  const _Float16* aRl = base + 1*TILEH;
  const _Float16* aIh = base + 2*TILEH;  const _Float16* aIl = base + 3*TILEH;
  const _Float16* bRh = base + 4*TILEH;  const _Float16* bRl = base + 5*TILEH;
  const _Float16* bIh = base + 6*TILEH;  const _Float16* bIl = base + 7*TILEH;

  // A frag: lane holds row m=wm+lr; lanes<16 K 0..7 & 16..23, lanes>=16 +8
  const int aoff = (wm + lr)*PITCH + hs*8;
  F16Frag fARh, fARl, fAIh, fAIl;
  fARh.h[0] = *(const v8h*)(aRh + aoff); fARh.h[1] = *(const v8h*)(aRh + aoff + 16);
  fARl.h[0] = *(const v8h*)(aRl + aoff); fARl.h[1] = *(const v8h*)(aRl + aoff + 16);
  fAIh.h[0] = *(const v8h*)(aIh + aoff); fAIh.h[1] = *(const v8h*)(aIh + aoff + 16);
  fAIl.h[0] = *(const v8h*)(aIl + aoff); fAIl.h[1] = *(const v8h*)(aIl + aoff + 16);

#pragma unroll
  for (int g = 0; g < 2; ++g) {
    // B frag: lane holds col n=wn+g*16+lr; lanes<16 K 0..7 & 16..23, lanes>=16 +8
    const int boff = (wn + g*16 + lr)*PITCH + hs*8;
    F16Frag fBRh, fBRl, fBIh, fBIl;
    fBRh.h[0] = *(const v8h*)(bRh + boff); fBRh.h[1] = *(const v8h*)(bRh + boff + 16);
    fBRl.h[0] = *(const v8h*)(bRl + boff); fBRl.h[1] = *(const v8h*)(bRl + boff + 16);
    fBIh.h[0] = *(const v8h*)(bIh + boff); fBIh.h[1] = *(const v8h*)(bIh + boff + 16);
    fBIl.h[0] = *(const v8h*)(bIl + boff); fBIl.h[1] = *(const v8h*)(bIl + boff + 16);

    // outR += (wR)^T R + (wI)^T I   (hi*hi + hi*lo + lo*hi each)
    accR[g]  = wmma_f16(fARh, fBRh, accR[g]);
    accR[g]  = wmma_f16(fARh, fBRl, accR[g]);
    accR[g]  = wmma_f16(fARl, fBRh, accR[g]);
    accR[g]  = wmma_f16(fAIh, fBIh, accR[g]);
    accR[g]  = wmma_f16(fAIh, fBIl, accR[g]);
    accR[g]  = wmma_f16(fAIl, fBIh, accR[g]);
    // ir = (wI)^T R
    accIp[g] = wmma_f16(fAIh, fBRh, accIp[g]);
    accIp[g] = wmma_f16(fAIh, fBRl, accIp[g]);
    accIp[g] = wmma_f16(fAIl, fBRh, accIp[g]);
    // ri = (wR)^T I  (no A/B NEG for f16 WMMA -> subtract in epilogue)
    accIm[g] = wmma_f16(fARh, fBIh, accIm[g]);
    accIm[g] = wmma_f16(fARh, fBIl, accIm[g]);
    accIm[g] = wmma_f16(fARl, fBIh, accIm[g]);
  }
}

__device__ __forceinline__ void epilogue(int b, int i0, int j0,
    int wm, int wn, int hs, int lr, v8f accR[2], v8f accIp[2], v8f accIm[2],
    float* __restrict__ outR, float* __restrict__ outI)
{
  const size_t ob = (size_t)b * (DIM*DIM);
#pragma unroll
  for (int g = 0; g < 2; ++g) {
    const int e = j0 + wn + g*16 + lr;
#pragma unroll
    for (int r = 0; r < 8; ++r) {
      const int d = i0 + wm + r + hs*8;
      const size_t o = ob + (size_t)d*DIM + e;
      outR[o] = accR[g][r];
      outI[o] = accIp[g][r] - accIm[g][r];
    }
  }
}

// ---------------------------------------------------------------------------
// TDM GEMM: each wave stages its own tile (wave w <-> ws array w) via
// TENSOR_LOAD_TO_LDS with built-in row padding; double-buffered LDS so the
// next K-step's DMA overlaps the current 24 WMMAs.
// ---------------------------------------------------------------------------
__global__ __launch_bounds__(256) void ComplexMixture_gemm_tdm(
    const _Float16* __restrict__ pk,
    float* __restrict__ outR, float* __restrict__ outI)
{
  __shared__ __align__(16) _Float16 sm[2*8*TILEH];   // 80 KB double-buffered

  const int t  = threadIdx.x;
  const int b  = blockIdx.z;
  const int i0 = blockIdx.y * 64;
  const int j0 = blockIdx.x * 64;

  const int w  = t >> 5;                 // wave 0..7 <-> tile/array 0..7
  const int L  = t & 31;
  const int wm = (w & 3) * 16;
  const int wn = (w >> 2) * 32;
  const int hs = L >> 4;
  const int lr = L & 15;

  const int rowBase = (w < 4) ? i0 : j0; // arrays 0..3 A-side, 4..7 B-side
  const unsigned long long arrBase = (unsigned long long)(uintptr_t)pk
      + 2ull*((unsigned long long)w*NELEM
              + ((unsigned long long)b*DIM + rowBase)*SEQ);
  const unsigned ldsBase = (unsigned)(uintptr_t)(void*)sm; // addr[31:0] = LDS offset

  // D# group1: data_size=4B, pad_enable, pad_interval=3 (16 dwords),
  // pad_amount=3 (4 dwords); tensor_dim0=256 dwords/row, stride=256;
  // tile = 16 dwords x 64 rows; 2D (tile_dim2=0); groups 2/3 zero.
  const v8i g1 = { (int)((2u<<16)|(1u<<20)|(3u<<22)|(3u<<25)),
                   (int)(256u << 16),    // tensor_dim0[15:0] in bits 63:48
                   (int)(6144u << 16),   // tensor_dim1[15:0] in bits 95:80
                   (int)(16u << 16),     // tile_dim0 in bits 127:112
                   64,                   // tile_dim1
                   256,                  // tensor_dim0_stride[31:0]
                   0, 0 };
  const v4i gz = {0, 0, 0, 0};
  const v8i gz8 = {0, 0, 0, 0, 0, 0, 0, 0};  // VADDR4 slot (unused per ISA)

  auto issue = [&](int buf, int s0) {
    unsigned long long ga = arrBase + (unsigned)(2*s0);
    v4u g0;
    g0[0] = 1u;                                              // count=1
    g0[1] = ldsBase + (unsigned)((buf*8 + w)*TILEH*2);       // lds_addr (bytes)
    g0[2] = (unsigned)ga;                                    // global_addr[31:0]
    g0[3] = ((unsigned)(ga >> 32) & 0x1FFFFFFu) | (2u << 30);// [56:32] | type=2
    __builtin_amdgcn_tensor_load_to_lds(g0, g1, gz, gz, gz8, 0);
  };

  v8f zero = {0,0,0,0,0,0,0,0};
  v8f accR[2]  = {zero, zero};
  v8f accIp[2] = {zero, zero};
  v8f accIm[2] = {zero, zero};

  issue(0, 0);                            // prefetch first K-slab
  for (int ks = 0; ks < SEQ/32; ++ks) {
    const int cur = ks & 1;
    __builtin_amdgcn_s_wait_tensorcnt(0); // my tile landed
    __syncthreads();                      // everyone's tiles landed
    if (ks + 1 < SEQ/32) issue(1 - cur, (ks + 1)*32);  // overlap next DMA
    compute_step(sm + cur*8*TILEH, wm, wn, hs, lr, accR, accIp, accIm);
  }
  epilogue(b, i0, j0, wm, wn, hs, lr, accR, accIp, accIm, outR, outI);
}

// ---------------------------------------------------------------------------
// Fallback (ws too small): inline f32->hi/lo conversion staging.
// ---------------------------------------------------------------------------
__device__ __forceinline__ void stage_f32(const float* __restrict__ src,
    const float* __restrict__ wt, int s0, int colBase,
    _Float16* tH, _Float16* tL, int t)
{
#pragma unroll
  for (int p = 0; p < 4; ++p) {
    int idx = t + p*256;
    int e   = idx & 63;
    int k   = (idx >> 6) * 2;
    float f0 = src[(s0 + k    )*DIM + colBase + e];
    float f1 = src[(s0 + k + 1)*DIM + colBase + e];
    if (wt) { f0 *= wt[s0 + k]; f1 *= wt[s0 + k + 1]; }
    unsigned p0 = pack_split(f0);
    unsigned p1 = pack_split(f1);
    *(unsigned*)(tH + e*PITCH + k) = (p0 & 0xFFFFu) | (p1 << 16);
    *(unsigned*)(tL + e*PITCH + k) = (p0 >> 16)     | (p1 & 0xFFFF0000u);
  }
}

__global__ __launch_bounds__(256) void ComplexMixture_gemm_fallback(
    const float* __restrict__ real, const float* __restrict__ imag,
    const float* __restrict__ weight,
    float* __restrict__ outR, float* __restrict__ outI)
{
  __shared__ __align__(16) _Float16 sm[8*TILEH];

  const int t  = threadIdx.x;
  const int b  = blockIdx.z;
  const int i0 = blockIdx.y * 64;
  const int j0 = blockIdx.x * 64;
  const int mb = b * (SEQ*DIM);

  const int w  = t >> 5;
  const int L  = t & 31;
  const int wm = (w & 3) * 16;
  const int wn = (w >> 2) * 32;
  const int hs = L >> 4;
  const int lr = L & 15;

  v8f zero = {0,0,0,0,0,0,0,0};
  v8f accR[2]  = {zero, zero};
  v8f accIp[2] = {zero, zero};
  v8f accIm[2] = {zero, zero};

  for (int ks = 0; ks < SEQ/32; ++ks) {
    const int s0 = ks * 32;
    __syncthreads();
    const float* wt = weight + b*SEQ;
    stage_f32(real + mb, wt,      s0, i0, sm + 0*TILEH, sm + 1*TILEH, t);
    stage_f32(imag + mb, wt,      s0, i0, sm + 2*TILEH, sm + 3*TILEH, t);
    stage_f32(real + mb, nullptr, s0, j0, sm + 4*TILEH, sm + 5*TILEH, t);
    stage_f32(imag + mb, nullptr, s0, j0, sm + 6*TILEH, sm + 7*TILEH, t);
    __syncthreads();
    compute_step(sm, wm, wn, hs, lr, accR, accIp, accIm);
  }
  epilogue(b, i0, j0, wm, wn, hs, lr, accR, accIp, accIm, outR, outI);
}

// ---------------------------------------------------------------------------
extern "C" void kernel_launch(void* const* d_in, const int* in_sizes, int n_in,
                              void* d_out, int out_size, void* d_ws, size_t ws_size,
                              hipStream_t stream)
{
  const float* real = (const float*)d_in[0];
  const float* imag = (const float*)d_in[1];
  const float* wt   = (const float*)d_in[2];
  float* outR = (float*)d_out;
  float* outI = outR + (size_t)BATCH*DIM*DIM;

  dim3 grid(DIM/64, DIM/64, BATCH);                 // 12 x 12 x 8
  const size_t need = (size_t)8 * NELEM * sizeof(_Float16);  // ~50 MB

  if (d_ws != nullptr && ws_size >= need) {
    dim3 tgrid(DIM/64, SEQ/64, BATCH);              // 12 x 8 x 8
    split_transpose_kernel<<<tgrid, 256, 0, stream>>>(real, imag, wt,
                                                      (unsigned*)d_ws);
    ComplexMixture_gemm_tdm<<<grid, 256, 0, stream>>>((const _Float16*)d_ws,
                                                      outR, outI);
  } else {
    ComplexMixture_gemm_fallback<<<grid, 256, 0, stream>>>(real, imag, wt,
                                                           outR, outI);
  }
}
